// Rollout_41291815583915
// MI455X (gfx1250) — matrix operand order repaired and checked by reading, not compiled, not run on hardware
//
#include <hip/hip_runtime.h>

#define VOCAB 32000
#define EMB   256
#define HID   512
#define SEQL  64
#define BATCH 32
#define GIVEN 16

typedef __attribute__((ext_vector_type(16))) __bf16 v16bf;
typedef __attribute__((ext_vector_type(8)))  __bf16 v8bf;
typedef __attribute__((ext_vector_type(8)))  float  v8f;

// ---------- helpers ----------
__device__ __forceinline__ unsigned short f2bf(float f) {
    unsigned int u = __float_as_uint(f);
    unsigned int r = (u + 0x7FFFu + ((u >> 16) & 1u)) >> 16;   // RNE
    return (unsigned short)r;
}

// Build A operand for v_wmma_f32_16x16x32_bf16.
// Lane L<16: row M=L, elems 0-7 = K ka..ka+7, elems 8-15 = K ka+16..ka+23 (ka = kb)
// Lane L>=16: same row M=L-16, ka = kb+8.
__device__ __forceinline__ v16bf load_a16(const unsigned short* p) {
    v8bf lo = *(const v8bf*)p;          // 16B
    v8bf hi = *(const v8bf*)(p + 16);   // 16B, +16 elements
    v16bf r;
#pragma unroll
    for (int i = 0; i < 8; ++i) { r[i] = lo[i]; r[8 + i] = hi[i]; }
    return r;
}

__device__ __forceinline__ unsigned int wang_hash(unsigned int s) {
    s = (s ^ 61u) ^ (s >> 16); s *= 9u; s ^= s >> 4; s *= 0x27d4eb2du; s ^= s >> 15;
    return s;
}

__device__ __forceinline__ float sigf(float x) { return 1.0f / (1.0f + __expf(-x)); }

// ---------- one-time conversions ----------
__global__ void convert_bf16_kernel(const float* __restrict__ src,
                                    unsigned short* __restrict__ dst, int n) {
    int i = blockIdx.x * blockDim.x + threadIdx.x;
    if (i < n) dst[i] = f2bf(src[i]);
}

__global__ void init_state_kernel(const float* __restrict__ h0, const float* __restrict__ c0,
                                  unsigned short* __restrict__ h_bf, float* __restrict__ c_state) {
    int i = blockIdx.x * blockDim.x + threadIdx.x;
    if (i < BATCH * HID) { h_bf[i] = f2bf(h0[i]); c_state[i] = c0[i]; }
}

// ---------- embedding gather ----------
__global__ void embed_sentence_kernel(const int* __restrict__ sentence, int t,
                                      const unsigned short* __restrict__ emb_bf,
                                      unsigned short* __restrict__ x_bf) {
    int b = blockIdx.x;                       // 32 blocks x 256 threads, EMB=256
    int tok = sentence[b * SEQL + t];
    x_bf[b * EMB + threadIdx.x] = emb_bf[(size_t)tok * EMB + threadIdx.x];
}

__global__ void embed_token_kernel(const int* __restrict__ tok,
                                   const unsigned short* __restrict__ emb_bf,
                                   unsigned short* __restrict__ x_bf) {
    int b = blockIdx.x;
    int tv = tok[b];
    x_bf[b * EMB + threadIdx.x] = emb_bf[(size_t)tv * EMB + threadIdx.x];
}

// ---------- fused LSTM cell step (gates GEMM via WMMA + elementwise) ----------
// grid.x = HID/32 = 16, block = 256 (8 waves).
// Wave w: gate = w&3, column sub-tile jtl = w>>2. Each wave computes BOTH batch
// M-tiles (rows 0-15 and 16-31) with two accumulators, loading B only once.
// gates[:, 0:512]=i, [512:1024]=f, [1024:1536]=g, [1536:2048]=o
__global__ __launch_bounds__(256) void lstm_step_kernel(
    const unsigned short* __restrict__ x_bf,   // [32][EMB]
    const unsigned short* __restrict__ h_in,   // [32][HID] bf16 (prev h)
    const unsigned short* __restrict__ Wi_bf,  // [2048][EMB]
    const unsigned short* __restrict__ Wh_bf,  // [2048][HID]
    const float* __restrict__ bi, const float* __restrict__ bh,
    float* __restrict__ c_state,               // [32][HID] f32, in-place
    unsigned short* __restrict__ h_out)        // [32][HID] bf16 (next h)
{
    __shared__ float lds_g[4][BATCH][32];      // gate x batch x 32-column slice
    const int tid   = threadIdx.x;
    const int wave  = tid >> 5;
    const int lane  = tid & 31;
    const int nloc  = lane & 15;
    const int hiL   = lane >> 4;               // 0 or 1
    const int gate  = wave & 3;                // 0..3
    const int jtl   = wave >> 2;               // 0/1: which 16-col sub-tile
    const int n     = gate * HID + blockIdx.x * 32 + jtl * 16 + nloc; // gate-matrix column

    v8f acc0, acc1;
    const float bias = bi[n] + bh[n];
#pragma unroll
    for (int r = 0; r < 8; ++r) { acc0[r] = bias; acc1[r] = bias; }

    const unsigned short* xrow0 = x_bf + nloc * EMB;          // M-tile 0 rows
    const unsigned short* xrow1 = x_bf + (16 + nloc) * EMB;   // M-tile 1 rows
    const unsigned short* hrow0 = h_in + nloc * HID;
    const unsigned short* hrow1 = h_in + (16 + nloc) * HID;
    const unsigned short* wiro  = Wi_bf + (size_t)n * EMB;
    const unsigned short* whro  = Wh_bf + (size_t)n * HID;

    // x @ Wi.T : K = EMB
#pragma unroll
    for (int kb = 0; kb < EMB; kb += 32) {
        v16bf B  = load_a16(wiro + kb + hiL * 8);             // loaded once, used twice
        v16bf A0 = load_a16(xrow0 + kb + hiL * 8);
        v16bf A1 = load_a16(xrow1 + kb + hiL * 8);
        acc0 = __builtin_amdgcn_wmma_f32_16x16x32_bf16(false, A0, false, B,
                                                       (short)0, acc0, false, false);
        acc1 = __builtin_amdgcn_wmma_f32_16x16x32_bf16(false, A1, false, B,
                                                       (short)0, acc1, false, false);
    }
    // h @ Wh.T : K = HID
#pragma unroll
    for (int kb = 0; kb < HID; kb += 32) {
        v16bf B  = load_a16(whro + kb + hiL * 8);
        v16bf A0 = load_a16(hrow0 + kb + hiL * 8);
        v16bf A1 = load_a16(hrow1 + kb + hiL * 8);
        acc0 = __builtin_amdgcn_wmma_f32_16x16x32_bf16(false, A0, false, B,
                                                       (short)0, acc0, false, false);
        acc1 = __builtin_amdgcn_wmma_f32_16x16x32_bf16(false, A1, false, B,
                                                       (short)0, acc1, false, false);
    }

    // spill tiles: lane L, elem r -> M = 8*hiL + r (+16 for acc1), Nloc = jtl*16 + (lane&15)
#pragma unroll
    for (int r = 0; r < 8; ++r) {
        lds_g[gate][hiL * 8 + r     ][jtl * 16 + nloc] = acc0[r];
        lds_g[gate][hiL * 8 + r + 16][jtl * 16 + nloc] = acc1[r];
    }
    __syncthreads();

    // fused elementwise LSTM update: 32*32 = 1024 elements, 4 per thread
    for (int e = tid; e < BATCH * 32; e += 256) {
        int mm = e >> 5, jl = e & 31;
        int j  = blockIdx.x * 32 + jl;
        float ig = lds_g[0][mm][jl];
        float fg = lds_g[1][mm][jl];
        float gg = lds_g[2][mm][jl];
        float og = lds_g[3][mm][jl];
        float cold = c_state[mm * HID + j];
        float c2 = sigf(fg) * cold + sigf(ig) * tanhf(gg);
        float h2 = sigf(og) * tanhf(c2);
        c_state[mm * HID + j] = c2;
        h_out[mm * HID + j]   = f2bf(h2);
    }
}

// ---------- tag projection: h(32x512) @ W_tag.T(512x32000) + b_tag ----------
// grid.x = VOCAB/128 = 250, block = 256 (8 waves). Wave w owns N-tile
// blockIdx.x*8 + w and computes BOTH batch M-tiles with two accumulators,
// halving W_tag L2 traffic (the binding resource of this kernel).
__global__ __launch_bounds__(256) void tag_gemm_kernel(
    const unsigned short* __restrict__ h_bf,   // [32][HID]
    const unsigned short* __restrict__ Wt_bf,  // [VOCAB][HID]
    const float* __restrict__ b_tag,
    float* __restrict__ tag_out,               // [32][64][VOCAB]
    int t)
{
    const int wave  = threadIdx.x >> 5;
    const int lane  = threadIdx.x & 31;
    const int nloc  = lane & 15;
    const int hiL   = lane >> 4;
    const int n     = (blockIdx.x * 8 + wave) * 16 + nloc;

    v8f acc0, acc1;
    const float bias = b_tag[n];
#pragma unroll
    for (int r = 0; r < 8; ++r) { acc0[r] = bias; acc1[r] = bias; }

    const unsigned short* hrow0 = h_bf + nloc * HID;
    const unsigned short* hrow1 = h_bf + (16 + nloc) * HID;
    const unsigned short* wrow  = Wt_bf + (size_t)n * HID;
    __builtin_prefetch(wrow + 256, 0, 1);   // global_prefetch_b8: 2nd half of weight row

#pragma unroll
    for (int kb = 0; kb < HID; kb += 32) {
        v16bf B  = load_a16(wrow + kb + hiL * 8);             // loaded once, used twice
        v16bf A0 = load_a16(hrow0 + kb + hiL * 8);
        v16bf A1 = load_a16(hrow1 + kb + hiL * 8);
        acc0 = __builtin_amdgcn_wmma_f32_16x16x32_bf16(false, A0, false, B,
                                                       (short)0, acc0, false, false);
        acc1 = __builtin_amdgcn_wmma_f32_16x16x32_bf16(false, A1, false, B,
                                                       (short)0, acc1, false, false);
    }

#pragma unroll
    for (int r = 0; r < 8; ++r) {
        int m0 = hiL * 8 + r;                                 // batch rows 0-15
        int m1 = m0 + 16;                                     // batch rows 16-31
        tag_out[((size_t)m0 * SEQL + t) * VOCAB + n] = acc0[r];
        tag_out[((size_t)m1 * SEQL + t) * VOCAB + n] = acc1[r];
    }
}

// ---------- categorical sampling via Gumbel-max over logits ----------
__global__ __launch_bounds__(256) void sample_kernel(const float* __restrict__ tag,
                                                     int t, int* __restrict__ tok,
                                                     float* __restrict__ ypred,
                                                     unsigned int seed) {
    __shared__ float sv[256];
    __shared__ int   si[256];
    const int b   = blockIdx.x;
    const int tid = threadIdx.x;
    const float* row = tag + ((size_t)b * SEQL + t) * VOCAB;
    float best = -3.4e38f; int bidx = 0;
    for (int nIdx = tid; nIdx < VOCAB; nIdx += 256) {
        unsigned int h = wang_hash(seed ^ (unsigned int)((t * BATCH + b) * VOCAB + nIdx));
        float u = (float)(h >> 8) * (1.0f / 16777216.0f) + 1e-12f;
        float g = -__logf(-__logf(u));
        float v = row[nIdx] + g;
        if (v > best) { best = v; bidx = nIdx; }
    }
    sv[tid] = best; si[tid] = bidx;
    __syncthreads();
    for (int s = 128; s > 0; s >>= 1) {
        if (tid < s && sv[tid + s] > sv[tid]) { sv[tid] = sv[tid + s]; si[tid] = si[tid + s]; }
        __syncthreads();
    }
    if (tid == 0) {
        tok[b] = si[0];
        if (ypred) ypred[b * SEQL + t] = (float)si[0];
    }
}

// ---------- y_pred for the given prefix ----------
__global__ void ypred_pre_kernel(const int* __restrict__ sentence, float* __restrict__ ypred) {
    int i = blockIdx.x * blockDim.x + threadIdx.x;   // 32*16
    if (i < BATCH * GIVEN) {
        int b = i >> 4, t = i & 15;
        ypred[b * SEQL + t] = (float)sentence[b * SEQL + t];
    }
}

// ---------- in-place log_softmax along vocab ----------
__global__ __launch_bounds__(256) void log_softmax_kernel(float* __restrict__ tag) {
    __shared__ float red[256];
    const int tid = threadIdx.x;
    float* p = tag + (size_t)blockIdx.x * VOCAB;     // 2048 rows
    float mx = -3.4e38f;
    for (int nIdx = tid; nIdx < VOCAB; nIdx += 256) mx = fmaxf(mx, p[nIdx]);
    red[tid] = mx; __syncthreads();
    for (int s = 128; s > 0; s >>= 1) { if (tid < s) red[tid] = fmaxf(red[tid], red[tid + s]); __syncthreads(); }
    mx = red[0]; __syncthreads();
    float sum = 0.0f;
    for (int nIdx = tid; nIdx < VOCAB; nIdx += 256) sum += __expf(p[nIdx] - mx);
    red[tid] = sum; __syncthreads();
    for (int s = 128; s > 0; s >>= 1) { if (tid < s) red[tid] += red[tid + s]; __syncthreads(); }
    float lse = mx + __logf(red[0]);
    for (int nIdx = tid; nIdx < VOCAB; nIdx += 256) p[nIdx] -= lse;
}

// ---------- host ----------
extern "C" void kernel_launch(void* const* d_in, const int* in_sizes, int n_in,
                              void* d_out, int out_size, void* d_ws, size_t ws_size,
                              hipStream_t stream) {
    const int*   sentence = (const int*)  d_in[0];
    const float* h0       = (const float*)d_in[1];
    const float* c0       = (const float*)d_in[2];
    /* given_num = 16 (compile-time)                 d_in[3] */
    const float* emb      = (const float*)d_in[4];
    const float* W_ih     = (const float*)d_in[5];
    const float* W_hh     = (const float*)d_in[6];
    const float* b_ih     = (const float*)d_in[7];
    const float* b_hh     = (const float*)d_in[8];
    const float* Wc_ih    = (const float*)d_in[9];
    const float* Wc_hh    = (const float*)d_in[10];
    const float* bc_ih    = (const float*)d_in[11];
    const float* bc_hh    = (const float*)d_in[12];
    const float* W_tag    = (const float*)d_in[13];
    const float* b_tag    = (const float*)d_in[14];

    float* tag_out = (float*)d_out;                              // [32][64][32000]
    float* ypred   = tag_out + (size_t)BATCH * SEQL * VOCAB;     // [32][64]

    // workspace carve-out (256B aligned slices)
    char* ws = (char*)d_ws;
    size_t off = 0;
    auto carve = [&](size_t bytes) -> char* {
        char* p = ws + off;
        off = (off + bytes + 255) & ~(size_t)255;
        return p;
    };
    unsigned short* Wih_bf  = (unsigned short*)carve((size_t)4 * HID * EMB * 2);
    unsigned short* Whh_bf  = (unsigned short*)carve((size_t)4 * HID * HID * 2);
    unsigned short* Wcih_bf = (unsigned short*)carve((size_t)4 * HID * EMB * 2);
    unsigned short* Wchh_bf = (unsigned short*)carve((size_t)4 * HID * HID * 2);
    unsigned short* Wtag_bf = (unsigned short*)carve((size_t)VOCAB * HID * 2);
    unsigned short* emb_bf  = (unsigned short*)carve((size_t)VOCAB * EMB * 2);
    unsigned short* h_bf0   = (unsigned short*)carve((size_t)BATCH * HID * 2);
    unsigned short* h_bf1   = (unsigned short*)carve((size_t)BATCH * HID * 2);
    unsigned short* x_bf    = (unsigned short*)carve((size_t)BATCH * EMB * 2);
    float*          c_state = (float*)        carve((size_t)BATCH * HID * 4);
    int*            tok     = (int*)          carve((size_t)BATCH * 4);

    // one-time f32 -> bf16 conversion of all GEMM operands
    auto cvt = [&](const float* s, unsigned short* d, int n) {
        convert_bf16_kernel<<<(n + 255) / 256, 256, 0, stream>>>(s, d, n);
    };
    cvt(W_ih,  Wih_bf,  4 * HID * EMB);
    cvt(W_hh,  Whh_bf,  4 * HID * HID);
    cvt(Wc_ih, Wcih_bf, 4 * HID * EMB);
    cvt(Wc_hh, Wchh_bf, 4 * HID * HID);
    cvt(W_tag, Wtag_bf, VOCAB * HID);
    cvt(emb,   emb_bf,  VOCAB * EMB);
    init_state_kernel<<<(BATCH * HID + 255) / 256, 256, 0, stream>>>(h0, c0, h_bf0, c_state);

    unsigned short* hin  = h_bf0;
    unsigned short* hout = h_bf1;

    for (int t = 0; t < SEQL; ++t) {
        if (t < GIVEN)
            embed_sentence_kernel<<<BATCH, EMB, 0, stream>>>(sentence, t, emb_bf, x_bf);
        else
            embed_token_kernel<<<BATCH, EMB, 0, stream>>>(tok, emb_bf, x_bf);

        if (t < GIVEN)
            lstm_step_kernel<<<HID / 32, 256, 0, stream>>>(x_bf, hin, Wih_bf, Whh_bf,
                                                           b_ih, b_hh, c_state, hout);
        else
            lstm_step_kernel<<<HID / 32, 256, 0, stream>>>(x_bf, hin, Wcih_bf, Wchh_bf,
                                                           bc_ih, bc_hh, c_state, hout);

        tag_gemm_kernel<<<VOCAB / 128, 256, 0, stream>>>(hout, Wtag_bf, b_tag, tag_out, t);

        if (t >= GIVEN - 1)   // sample at t=15 (feeds rollout) and every rollout step
            sample_kernel<<<BATCH, 256, 0, stream>>>(tag_out, t, tok,
                                                     (t >= GIVEN) ? ypred : nullptr, 42u);

        unsigned short* tmp = hin; hin = hout; hout = tmp;
    }

    ypred_pre_kernel<<<(BATCH * GIVEN + 255) / 256, 256, 0, stream>>>(sentence, ypred);
    log_softmax_kernel<<<BATCH * SEQL, 256, 0, stream>>>(tag_out);
}
